// NonLinearDipoleReadoutBlock_18476949307845
// MI455X (gfx1250) — matrix operand order repaired and checked by reading, not compiled, not run on hardware
//
#include <hip/hip_runtime.h>

typedef __attribute__((ext_vector_type(16))) _Float16 v16h;
typedef __attribute__((ext_vector_type(8)))  float    v8f;

#define XSTRIDE    4608
#define N_ROWS     32768
#define N_TILES    2048          /* 16 rows per tile */
#define SW         520           /* padded f16 weight row stride (halves) */
#define WT_HALVES  (2 * 128 * SW)
#define WT_BYTES   (WT_HALVES * 2)            /* 266240 */
#define NPART      938                        /* ceil(240001/256) trapz blocks */
#define TLDS_OFF   WT_BYTES
#define OACC_OFF   (WT_BYTES + 16 * 128 * 4)  /* 274432 */
#define SMEM_BYTES (OACC_OFF + 48 * 4)        /* 274624 <= 320KB */
#define INV_IN     0.04419417382415922f       /* 1/sqrt(512) */
#define INV_H      0.08838834764831843f       /* 1/sqrt(128) */

union Frag {
  v16h     v;
  _Float16 h[16];
  uint4    u[2];
};

// ---------------------------------------------------------------------------
// Kernel 1: weight transpose/convert to f16 (padded, B-fragment friendly) and
// deterministic partial sums of trapz(tanh(z)^2 * pdf(z)).
// ---------------------------------------------------------------------------
__global__ void prep_kernel(const float* __restrict__ W0, const float* __restrict__ W1,
                            _Float16* __restrict__ wt, float* __restrict__ partials) {
  __shared__ float red[256];
  const int b = blockIdx.x, t = threadIdx.x;
  if (b < 512) {
    const int id  = b * 256 + t;        // 0..131071
    const int mat = id >> 16;           // 0 -> W0 slice, 1 -> W1
    const int r   = id & 65535;
    const int h   = r >> 9;             // 0..127
    const int c   = r & 511;            // 0..511
    const float v = (mat == 0) ? W0[c * 384 + 128 + h] : W1[c * 128 + h];
    wt[(mat * 128 + h) * SW + c] = (_Float16)v;
  } else {
    const int i = (b - 512) * 256 + t;
    float f = 0.0f;
    if (i <= 240000) {
      const double z  = -12.0 + 1.0e-4 * (double)i;
      const double w  = (i == 0 || i == 240000) ? 0.5 : 1.0;
      const double th = tanh(z);
      f = (float)(w * th * th * exp(-0.5 * z * z) * 0.3989422804014326779);
    }
    red[t] = f;
    __syncthreads();
    #pragma unroll
    for (int s = 128; s > 0; s >>= 1) {
      if (t < s) red[t] += red[t + s];
      __syncthreads();
    }
    if (t == 0) partials[b - 512] = red[0];
  }
}

// ---------------------------------------------------------------------------
// Kernel 2: deterministic reduce -> TANH_C = 1/sqrt(integral)
// ---------------------------------------------------------------------------
__global__ void reduce_tanh_kernel(const float* __restrict__ partials, float* __restrict__ out_c) {
  __shared__ float red[256];
  const int t = threadIdx.x;
  float s = 0.0f;
  for (int i = t; i < NPART; i += 256) s += partials[i];
  red[t] = s;
  __syncthreads();
  #pragma unroll
  for (int st = 128; st > 0; st >>= 1) {
    if (t < st) red[t] += red[t + st];
    __syncthreads();
  }
  if (t == 0) out_c[0] = (float)(1.0 / sqrt((double)red[0] * 1.0e-4));
}

// ---------------------------------------------------------------------------
// Kernel 3: fused GEMMs (f16 WMMA, fp32 accum) + tanh gating + h-contraction.
// 8 waves: wave = (gemm in 0..3) x (h-half in 0..1). gemm0 -> g1, gemm1..3 -> y1[m].
// ---------------------------------------------------------------------------
__global__ void __launch_bounds__(256, 1)
fused_kernel(const float* __restrict__ x, const float* __restrict__ w_out,
             const uint4* __restrict__ wt_g, const float* __restrict__ tanhc_p,
             float* __restrict__ out) {
  extern __shared__ char smem[];
  _Float16* wt    = (_Float16*)smem;
  float*    t_lds = (float*)(smem + TLDS_OFF);
  float*    oacc  = (float*)(smem + OACC_OFF);

  const int tid = threadIdx.x;

  // --- Stage f16 weights into LDS with CDNA5 async global->LDS DMA. -------
  // No static LDS in this kernel, so dynamic LDS starts at offset 0; the
  // low 32 bits of a generic LDS pointer are the LDS byte address (ISA 10.2).
  {
    const unsigned       lds0 = (unsigned)(size_t)smem;
    const unsigned long long g0 = (unsigned long long)wt_g;
    for (int i = tid; i < WT_BYTES / 16; i += 256) {
      const unsigned           lo = lds0 + (unsigned)i * 16u;
      const unsigned long long ga = g0 + (unsigned long long)i * 16ull;
      asm volatile("global_load_async_to_lds_b128 %0, %1, off"
                   :: "v"(lo), "v"(ga) : "memory");
    }
    asm volatile("s_wait_asynccnt 0x0" ::: "memory");
  }
  __syncthreads();

  const int wid  = __builtin_amdgcn_readfirstlane(tid >> 5);  // wave-uniform -> SGPR
  const int lane = tid & 31;
  const int ln   = lane & 15;   // A: row-in-tile, B: column h, C/D: column h
  const int hi   = lane >> 4;   // K-group select / row+8 select
  const int gemm = wid & 3;     // 0 = gating GEMM, 1..3 = y1 m-GEMMs
  const int hh   = wid >> 2;    // h half: 0 -> h 0..63, 1 -> h 64..127
  const int wsel = (gemm == 0) ? 0 : 1;
  const int mcol = gemm - 1;
  const float scaleT = tanhc_p[0] * INV_H * INV_IN;

  const _Float16* bbase = wt + (wsel * 128 + hh * 64 + ln) * SW + 8 * hi;

  for (int t = blockIdx.x; t < N_TILES; t += gridDim.x) {
    const int r0 = t * 16;
    if (tid < 48) oacc[tid] = 0.0f;

    const float* xrow = x + (size_t)(r0 + ln) * XSTRIDE;
    v8f acc[4] = {};

    // Load 4 B fragments into distinct regs, then 4 back-to-back WMMAs.
    auto mm = [&](const Frag& a, int k0) {
      Frag b[4];
      #pragma unroll
      for (int i = 0; i < 4; ++i) {
        const uint4* bp = (const uint4*)(bbase + i * (16 * SW) + k0);
        b[i].u[0] = bp[0];
        b[i].u[1] = bp[2];
      }
      #pragma unroll
      for (int i = 0; i < 4; ++i)
        acc[i] = __builtin_amdgcn_wmma_f32_16x16x32_f16(
            false, a.v, false, b[i].v, (short)0, acc[i], false, false);
    };

    if (gemm == 0) {
      // x0 GEMM: contiguous K; lane holds K = k0+kb+[0..7], k0+16+kb+[0..7].
      auto cvt4 = [](float4 c0, float4 c1, float4 c2, float4 c3) {
        Frag a;
        a.h[0]  = (_Float16)c0.x; a.h[1]  = (_Float16)c0.y;
        a.h[2]  = (_Float16)c0.z; a.h[3]  = (_Float16)c0.w;
        a.h[4]  = (_Float16)c1.x; a.h[5]  = (_Float16)c1.y;
        a.h[6]  = (_Float16)c1.z; a.h[7]  = (_Float16)c1.w;
        a.h[8]  = (_Float16)c2.x; a.h[9]  = (_Float16)c2.y;
        a.h[10] = (_Float16)c2.z; a.h[11] = (_Float16)c2.w;
        a.h[12] = (_Float16)c3.x; a.h[13] = (_Float16)c3.y;
        a.h[14] = (_Float16)c3.z; a.h[15] = (_Float16)c3.w;
        return a;
      };
      const float* pa = xrow + 8 * hi;
      float4 c0 = *(const float4*)(pa);
      float4 c1 = *(const float4*)(pa + 4);
      float4 c2 = *(const float4*)(pa + 16);
      float4 c3 = *(const float4*)(pa + 20);
      #pragma unroll 5
      for (int step = 0; step < 15; ++step) {        // software-pipelined
        Frag a = cvt4(c0, c1, c2, c3);
        const float* pn = pa + (step + 1) * 32;
        c0 = *(const float4*)(pn);
        c1 = *(const float4*)(pn + 4);
        c2 = *(const float4*)(pn + 16);
        c3 = *(const float4*)(pn + 20);
        mm(a, step * 32);
      }
      Frag a = cvt4(c0, c1, c2, c3);
      mm(a, 480);
    } else {
      // y1 GEMMs: x1[n,c,m] = x[n, 512 + 3c + m], stride-3 gather.
      auto cvt16 = [](const float (&r)[16]) {
        Frag a;
        #pragma unroll
        for (int j = 0; j < 16; ++j) a.h[j] = (_Float16)r[j];
        return a;
      };
      const float* pa = xrow + 512 + mcol + 24 * hi;  // 3*(8*hi)
      float r[16];
      #pragma unroll
      for (int j = 0; j < 8; ++j) { r[j] = pa[3 * j]; r[8 + j] = pa[48 + 3 * j]; }
      #pragma unroll 5
      for (int step = 0; step < 15; ++step) {        // software-pipelined
        Frag a = cvt16(r);
        const float* pn = pa + (step + 1) * 96;      // 3*32 floats per K-step
        #pragma unroll
        for (int j = 0; j < 8; ++j) { r[j] = pn[3 * j]; r[8 + j] = pn[48 + 3 * j]; }
        mm(a, step * 32);
      }
      Frag a = cvt16(r);
      mm(a, 480);
    }

    if (gemm == 0) {
      // t[n,h] = w_out[h] * TANH_C*inv_h*inv_in * tanh(inv_in * G_raw[n,h])
      #pragma unroll
      for (int i = 0; i < 4; ++i) {
        const int h = hh * 64 + i * 16 + ln;
        const float wo = w_out[h] * scaleT;
        #pragma unroll
        for (int v = 0; v < 8; ++v) {
          const int M = v + 8 * hi;
          t_lds[M * 128 + h] = wo * tanhf(INV_IN * acc[i][v]);
        }
      }
    }
    __syncthreads();
    if (gemm != 0) {
      // out[n,m] = sum_h t[n,h] * Yraw_m[n,h]
      #pragma unroll
      for (int v = 0; v < 8; ++v) {
        const int M = v + 8 * hi;
        float val = 0.0f;
        #pragma unroll
        for (int i = 0; i < 4; ++i) {
          const int h = hh * 64 + i * 16 + ln;
          val += acc[i][v] * t_lds[M * 128 + h];
        }
        #pragma unroll
        for (int o = 1; o < 16; o <<= 1) val += __shfl_xor(val, o, 32);
        if (ln == 0) atomicAdd(&oacc[mcol * 16 + M], val);  // 2 adds/addr: commutative -> deterministic
      }
    }
    __syncthreads();
    if (tid < 48) {
      const int row = tid & 15, m = tid >> 4;
      out[(size_t)(r0 + row) * 3 + m] = oacc[m * 16 + row];
    }
  }
}

// ---------------------------------------------------------------------------
extern "C" void kernel_launch(void* const* d_in, const int* in_sizes, int n_in,
                              void* d_out, int out_size, void* d_ws, size_t ws_size,
                              hipStream_t stream) {
  (void)in_sizes; (void)n_in; (void)out_size; (void)ws_size;
  const float* x     = (const float*)d_in[0];
  const float* W0    = (const float*)d_in[1];
  const float* W1    = (const float*)d_in[2];
  const float* w_out = (const float*)d_in[4];   // W2 (d_in[3]) is dead code
  float* out = (float*)d_out;

  char*      ws       = (char*)d_ws;
  _Float16*  wt       = (_Float16*)ws;
  float*     partials = (float*)(ws + WT_BYTES);
  float*     tanhc    = partials + NPART;

  prep_kernel<<<512 + NPART, 256, 0, stream>>>(W0, W1, wt, partials);
  reduce_tanh_kernel<<<1, 256, 0, stream>>>(partials, tanhc);
  fused_kernel<<<512, 256, SMEM_BYTES, stream>>>(x, w_out, (const uint4*)ws, tanhc, out);
}
